// MambaBlock_27152783245824
// MI455X (gfx1250) — matrix operand order repaired and checked by reading, not compiled, not run on hardware
//
#include <hip/hip_runtime.h>
#include <hip/hip_bf16.h>
#include <math.h>

typedef __attribute__((ext_vector_type(16))) __bf16 v16bf;
typedef __attribute__((ext_vector_type(8)))  __bf16 bf16x8;
typedef __attribute__((ext_vector_type(8)))  float  v8f;

#define BB 2
#define LL 2048
#define DM 1024
#define DI 2048
#define DS 16
#define DTR 64
#define ML (BB*LL)          // 4096 rows in all GEMMs

// async copy: INST_OFFSET is added to BOTH the LDS and the global address,
// so one (ldsaddr, voff) pair serves a whole 128B row via offset:0..112.
#define ASYNC_CP(ldsaddr, voff, sbase, IMM)                                  \
    asm volatile("global_load_async_to_lds_b128 %0, %1, %2 offset:" IMM     \
                 :: "v"(ldsaddr), "v"(voff), "s"(sbase) : "memory")
#define WAIT_ASYNC(IMM) asm volatile("s_wait_asynccnt " IMM ::: "memory")

// ---------------------------------------------------------------------------
// Conversion kernels (one-shot, tiny vs GEMM cost)
// ---------------------------------------------------------------------------
__global__ __launch_bounds__(256) void k_f32_to_bf16(const float* __restrict__ in,
                                                     __bf16* __restrict__ out, int n) {
    int i = blockIdx.x * 256 + threadIdx.x;
    if (i < n) out[i] = (__bf16)in[i];
}

// in: rows x cols (row-major f32) -> out: cols x rows (row-major bf16) == in^T
__global__ __launch_bounds__(256) void k_f32_to_bf16_T(const float* __restrict__ in,
                                                       __bf16* __restrict__ out,
                                                       int rows, int cols) {
    int i = blockIdx.x * 256 + threadIdx.x;
    if (i < rows * cols) {
        int r = i / cols, c = i % cols;
        out[(size_t)c * rows + r] = (__bf16)in[i];
    }
}

__device__ __forceinline__ v16bf make_frag(bf16x8 lo, bf16x8 hi) {
    v16bf f;
    #pragma unroll
    for (int i = 0; i < 8; ++i) { f[i] = lo[i]; f[8 + i] = hi[i]; }
    return f;
}

// ---------------------------------------------------------------------------
// Fully async-staged bf16 WMMA GEMM:  C[M,N] = A[M,K] @ B[K,N],  Bt = B^T.
// Block: 256 thr = 8 waves; block tile 256(M) x 64(N); wave tile 32 x 64.
// Per K-step (64): A tile 256x64 (32KB) + B tile 64x64 (8KB) staged to LDS via
// global_load_async_to_lds_b128, double-buffered (80KB LDS total, ASYNCcnt),
// consumed via ds_load_b128; 16 WMMA per wave between barriers.
// EPI==1: C = softplus(acc + bias[col]).
// Requires: M%256==0, N%64==0, K%64==0. No early exits -> EXEC all-ones.
// ---------------------------------------------------------------------------
template<int EPI>
__global__ __launch_bounds__(256)
void k_gemm_bf16_lds(const __bf16* __restrict__ A, const __bf16* __restrict__ Bt,
                     float* __restrict__ C, const float* __restrict__ bias,
                     int M, int N, int K)
{
    __shared__ __bf16 shA[2][256 * 64];   // [buf][row*64 + kk]   2 x 32KB
    __shared__ __bf16 shB[2][64 * 64];    // [buf][col*64 + kk]   2 x 8KB

    const int tid  = threadIdx.x;
    const int lane = tid & 31;
    const int w    = tid >> 5;
    const int l16  = lane & 15;
    const int hi   = lane >> 4;

    const int rowBlks = M >> 8;
    const int rowBlk  = blockIdx.x % rowBlks;
    const int colBlk  = blockIdx.x / rowBlks;
    const int colBase = colBlk * 64;
    const int rowBase = rowBlk * 256;
    const int rbase   = rowBase + w * 32;

    // ---- per-thread async-copy addressing -------------------------------
    // A: thread t stages row t of the block tile (128B = 8 chunks via offset)
    const unsigned shABase = (unsigned)(unsigned long long)(const void*)(&shA[0][0]);
    const unsigned shBBase = (unsigned)(unsigned long long)(const void*)(&shB[0][0]);
    const unsigned voffA  = (unsigned)(tid * K * 2);
    const unsigned ldsA0  = shABase + (unsigned)(tid * 128);
    // B: thread t stages chunks t and t+256 (cols 0-31 / 32-63)
    const int bcol = tid >> 3, bq = tid & 7;
    const unsigned voffB0 = (unsigned)((bcol * K + bq * 8) * 2);
    const unsigned voffB1 = voffB0 + (unsigned)(32 * K * 2);
    const unsigned ldsB0  = shBBase + (unsigned)(bcol * 128 + bq * 16);
    const unsigned long long gA0 = (unsigned long long)(const void*)(A  + (size_t)rowBase * K);
    const unsigned long long gB0 = (unsigned long long)(const void*)(Bt + (size_t)colBase * K);

    auto issue_tile = [&](int buf, int kstep) {
        const unsigned long long gA = gA0 + (unsigned long long)kstep * 128;
        const unsigned long long gB = gB0 + (unsigned long long)kstep * 128;
        const unsigned la  = ldsA0 + (unsigned)(buf * 32768);
        const unsigned lb0 = ldsB0 + (unsigned)(buf * 8192);
        const unsigned lb1 = lb0 + 4096u;
        ASYNC_CP(la, voffA, gA, "0");   ASYNC_CP(la, voffA, gA, "16");
        ASYNC_CP(la, voffA, gA, "32");  ASYNC_CP(la, voffA, gA, "48");
        ASYNC_CP(la, voffA, gA, "64");  ASYNC_CP(la, voffA, gA, "80");
        ASYNC_CP(la, voffA, gA, "96");  ASYNC_CP(la, voffA, gA, "112");
        ASYNC_CP(lb0, voffB0, gB, "0");
        ASYNC_CP(lb1, voffB1, gB, "0");
    };

    v8f acc[2][4];
    #pragma unroll
    for (int m = 0; m < 2; ++m)
        #pragma unroll
        for (int t = 0; t < 4; ++t)
            #pragma unroll
            for (int r = 0; r < 8; ++r) acc[m][t][r] = 0.0f;

    auto compute = [&](int buf) {
        const __bf16* sa = &shA[buf][0];
        const __bf16* sb = &shB[buf][0];
        #pragma unroll
        for (int s = 0; s < 2; ++s) {                    // two 32-K slices
            // A frags: lanes 0-15 K{0..7,16..23}, lanes 16-31 K{8..15,24..31}
            const __bf16* ap0 = sa + (w * 32 + l16) * 64 + s * 32 + hi * 8;
            const __bf16* ap1 = ap0 + 16 * 64;
            v16bf af0 = make_frag(*(const bf16x8*)(ap0), *(const bf16x8*)(ap0 + 16));
            v16bf af1 = make_frag(*(const bf16x8*)(ap1), *(const bf16x8*)(ap1 + 16));
            #pragma unroll
            for (int t = 0; t < 4; ++t) {
                // B frag: lane = col n (t*16+l16), K{hi*8..hi*8+7, 16+hi*8..16+hi*8+7}
                const __bf16* bp = sb + (t * 16 + l16) * 64 + s * 32 + hi * 8;
                v16bf bfrag = make_frag(*(const bf16x8*)(bp), *(const bf16x8*)(bp + 16));
                acc[0][t] = __builtin_amdgcn_wmma_f32_16x16x32_bf16(
                                false, af0, false, bfrag, (short)0, acc[0][t], false, false);
                acc[1][t] = __builtin_amdgcn_wmma_f32_16x16x32_bf16(
                                false, af1, false, bfrag, (short)0, acc[1][t], false, false);
            }
        }
    };

    const int steps = K >> 6;

    issue_tile(0, 0);                                   // prologue
    for (int k = 0; k < steps - 1; ++k) {               // branch-free steady state
        issue_tile((k + 1) & 1, k + 1);
        WAIT_ASYNC("0xa");                              // prev tile (10 asyncs) done
        __syncthreads();
        compute(k & 1);
        __syncthreads();                                // reads done before overwrite
    }
    WAIT_ASYNC("0x0");
    __syncthreads();
    compute((steps - 1) & 1);

    // C/D layout: VGPR r -> row = base + hi*8 + r, col = colBase + t*16 + l16
    #pragma unroll
    for (int m = 0; m < 2; ++m) {
        #pragma unroll
        for (int t = 0; t < 4; ++t) {
            const int col = colBase + t * 16 + l16;
            float* crow = C + (size_t)(rbase + m * 16 + hi * 8) * N + col;
            float bv = (EPI == 1) ? bias[col] : 0.0f;
            #pragma unroll
            for (int r = 0; r < 8; ++r) {
                float v = acc[m][t][r];
                if (EPI == 1) {
                    v += bv;
                    v = (v > 20.0f) ? v : log1pf(__expf(v));   // softplus
                }
                crow[(size_t)r * N] = v;
            }
        }
    }
}

// ---------------------------------------------------------------------------
// Direct-global WMMA GEMM for narrow N (GEMM2, N=96). Wave: 16 x 32 strip.
// ---------------------------------------------------------------------------
template<int NT>
__global__ __launch_bounds__(256)
void k_gemm_bf16_g(const __bf16* __restrict__ A, const __bf16* __restrict__ Bt,
                   float* __restrict__ C, int M, int N, int K)
{
    const int lane = threadIdx.x & 31;
    const int gwid = blockIdx.x * 8 + (threadIdx.x >> 5);
    const int tiles_m   = M >> 4;
    const int tn_groups = N / (16 * NT);
    const int tm = gwid % tiles_m;
    const int tg = gwid / tiles_m;
    if (tg >= tn_groups) return;              // wave-uniform exit

    const int l16 = lane & 15;
    const int hi  = lane >> 4;

    v8f acc[NT];
    #pragma unroll
    for (int t = 0; t < NT; ++t)
        #pragma unroll
        for (int r = 0; r < 8; ++r) acc[t][r] = 0.0f;

    const __bf16* __restrict__ arow = A + (size_t)(tm * 16 + l16) * K;

    for (int k0 = 0; k0 < K; k0 += 32) {
        v16bf afrag = make_frag(*(const bf16x8*)(arow + k0 + hi * 8),
                                *(const bf16x8*)(arow + k0 + 16 + hi * 8));
        #pragma unroll
        for (int t = 0; t < NT; ++t) {
            const int col = tg * (16 * NT) + t * 16 + l16;
            const __bf16* brow = Bt + (size_t)col * K;
            v16bf bfrag = make_frag(*(const bf16x8*)(brow + k0 + hi * 8),
                                    *(const bf16x8*)(brow + k0 + 16 + hi * 8));
            acc[t] = __builtin_amdgcn_wmma_f32_16x16x32_bf16(
                         false, afrag, false, bfrag, (short)0, acc[t], false, false);
        }
    }

    #pragma unroll
    for (int t = 0; t < NT; ++t) {
        const int col = tg * (16 * NT) + t * 16 + l16;
        float* crow = C + (size_t)(tm * 16 + hi * 8) * N + col;
        #pragma unroll
        for (int r = 0; r < 8; ++r) crow[(size_t)r * N] = acc[t][r];
    }
}

// ---------------------------------------------------------------------------
// Causal depthwise conv (K=4) + SiLU -> u (f32) and u (bf16 for next GEMM)
// ---------------------------------------------------------------------------
__global__ __launch_bounds__(256)
void k_conv_silu(const float* __restrict__ xz, const float* __restrict__ Wc,
                 const float* __restrict__ bc, float* __restrict__ u,
                 __bf16* __restrict__ ub)
{
    int i = blockIdx.x * 256 + threadIdx.x;           // over B*L*DI
    if (i >= ML * DI) return;
    int d  = i % DI;
    int bl = i / DI;
    int l  = bl % LL;
    int b  = bl / LL;
    float s = bc[d];
    #pragma unroll
    for (int t = 0; t < 4; ++t) {
        int ls = l - 3 + t;
        if (ls >= 0) s += xz[((size_t)(b * LL + ls)) * (2 * DI) + d] * Wc[t * DI + d];
    }
    float su = s / (1.0f + __expf(-s));               // silu
    u[i]  = su;
    ub[i] = (__bf16)su;
}

// dbc: (B*L) x 96 ; take cols [0,64) as bf16 for the delta GEMM
__global__ __launch_bounds__(256)
void k_extract_din(const float* __restrict__ dbc, __bf16* __restrict__ din) {
    int i = blockIdx.x * 256 + threadIdx.x;           // ML*64
    if (i >= ML * DTR) return;
    int m = i >> 6, k = i & 63;
    din[i] = (__bf16)dbc[(size_t)m * 96 + k];
}

// ---------------------------------------------------------------------------
// Selective scan: one thread per (b, d) channel; 16 states in registers.
// Fuses  y = scan + u*D,  gate = silu(res),  out = bf16(y*gate).
// ---------------------------------------------------------------------------
__global__ __launch_bounds__(256)
void k_scan(const float* __restrict__ delta, const float* __restrict__ u,
            const float* __restrict__ dbc, const float* __restrict__ xz,
            const float* __restrict__ A_log, const float* __restrict__ D_param,
            __bf16* __restrict__ yg)
{
    int tid = blockIdx.x * 256 + threadIdx.x;
    if (tid >= BB * DI) return;
    int d = tid % DI;
    int b = tid / DI;

    float Av[DS], st[DS];
    #pragma unroll
    for (int n = 0; n < DS; ++n) { Av[n] = -__expf(A_log[d * DS + n]); st[n] = 0.0f; }
    float Dp = D_param[d];

    for (int l = 0; l < LL; ++l) {
        size_t bl = (size_t)(b * LL + l);
        float dl = delta[bl * DI + d];
        float uu = u[bl * DI + d];
        const float* bcrow = dbc + bl * 96;
        float dlu = dl * uu;
        float y = 0.0f;
        #pragma unroll
        for (int n = 0; n < DS; ++n) {
            float dA = __expf(dl * Av[n]);
            st[n] = dA * st[n] + dlu * bcrow[64 + n];   // B part
            y += st[n] * bcrow[80 + n];                 // C part
        }
        y += uu * Dp;
        float r  = xz[bl * (2 * DI) + DI + d];          // res
        float ga = r / (1.0f + __expf(-r));             // silu(res)
        yg[bl * DI + d] = (__bf16)(y * ga);
    }
}

// ---------------------------------------------------------------------------
// Host launch
// ---------------------------------------------------------------------------
static inline int gemm_grid_g(int M, int N, int NT) {
    int waves = (M / 16) * (N / (16 * NT));
    return (waves + 7) / 8;
}

extern "C" void kernel_launch(void* const* d_in, const int* in_sizes, int n_in,
                              void* d_out, int out_size, void* d_ws, size_t ws_size,
                              hipStream_t stream) {
    const float* x       = (const float*)d_in[0];
    const float* W_in    = (const float*)d_in[1];   // (1024, 4096)
    const float* W_conv  = (const float*)d_in[2];   // (4, 2048)
    const float* b_conv  = (const float*)d_in[3];
    const float* W_xproj = (const float*)d_in[4];   // (2048, 96)
    const float* W_dt    = (const float*)d_in[5];   // (64, 2048)
    const float* b_dt    = (const float*)d_in[6];
    const float* A_log   = (const float*)d_in[7];   // (2048, 16)
    const float* D_param = (const float*)d_in[8];
    const float* W_out   = (const float*)d_in[9];   // (2048, 1024)
    float* out = (float*)d_out;

    char* ws = (char*)d_ws;
    size_t o = 0;
    auto alloc = [&](size_t bytes) { char* p = ws + o; o += (bytes + 255) & ~(size_t)255; return p; };
    __bf16* Wib  = (__bf16*)alloc((size_t)DM * 2 * DI * 2);      // W_in^T  (4096 x 1024)
    __bf16* xb   = (__bf16*)alloc((size_t)ML * DM * 2);          // x bf16
    float*  xz   = (float*) alloc((size_t)ML * 2 * DI * 4);      // x_and_res
    float*  u    = (float*) alloc((size_t)ML * DI * 4);
    __bf16* ub   = (__bf16*)alloc((size_t)ML * DI * 2);
    __bf16* Wxb  = (__bf16*)alloc((size_t)96 * DI * 2);          // W_xproj^T (96 x 2048)
    float*  dbc  = (float*) alloc((size_t)ML * 96 * 4);
    __bf16* din  = (__bf16*)alloc((size_t)ML * DTR * 2);
    __bf16* Wdtb = (__bf16*)alloc((size_t)DI * DTR * 2);         // W_dt^T (2048 x 64)
    float*  dlt  = (float*) alloc((size_t)ML * DI * 4);
    __bf16* Wob  = (__bf16*)alloc((size_t)DM * DI * 2);          // W_out^T (1024 x 2048)
    __bf16* yg   = (__bf16*)alloc((size_t)ML * DI * 2);
    (void)ws_size;

    // weight / activation conversions
    k_f32_to_bf16_T<<<(DM * 2 * DI + 255) / 256, 256, 0, stream>>>(W_in, Wib, DM, 2 * DI);
    k_f32_to_bf16  <<<(ML * DM     + 255) / 256, 256, 0, stream>>>(x, xb, ML * DM);
    k_f32_to_bf16_T<<<(DI * 96     + 255) / 256, 256, 0, stream>>>(W_xproj, Wxb, DI, 96);
    k_f32_to_bf16_T<<<(DTR * DI    + 255) / 256, 256, 0, stream>>>(W_dt, Wdtb, DTR, DI);
    k_f32_to_bf16_T<<<(DI * DM     + 255) / 256, 256, 0, stream>>>(W_out, Wob, DI, DM);

    // GEMM1: xz = x @ W_in   (4096 x 4096, K=1024)  [async LDS-pipelined]
    k_gemm_bf16_lds<0><<<(ML / 256) * (2 * DI / 64), 256, 0, stream>>>(
        xb, Wib, xz, nullptr, ML, 2 * DI, DM);

    // depthwise causal conv + SiLU -> u
    k_conv_silu<<<(ML * DI + 255) / 256, 256, 0, stream>>>(xz, W_conv, b_conv, u, ub);

    // GEMM2: dbc = u @ W_xproj   (4096 x 96, K=2048)  [direct-global, narrow N]
    k_gemm_bf16_g<2><<<gemm_grid_g(ML, 96, 2), 256, 0, stream>>>(
        ub, Wxb, dbc, ML, 96, DI);

    // delta = softplus(dbc[:, :64] @ W_dt + b_dt)   (4096 x 2048, K=64)
    k_extract_din<<<(ML * DTR + 255) / 256, 256, 0, stream>>>(dbc, din);
    k_gemm_bf16_lds<1><<<(ML / 256) * (DI / 64), 256, 0, stream>>>(
        din, Wdtb, dlt, b_dt, ML, DI, DTR);

    // selective scan + u*D + silu(res) gating -> yg (bf16)
    k_scan<<<(BB * DI + 255) / 256, 256, 0, stream>>>(dlt, u, dbc, xz, A_log, D_param, yg);

    // GEMM4: out = yg @ W_out   (4096 x 1024, K=2048)  [async LDS-pipelined]
    k_gemm_bf16_lds<0><<<(ML / 256) * (DM / 64), 256, 0, stream>>>(
        yg, Wob, out, nullptr, ML, DM, DI);
}
